// Net_85796266705242
// MI455X (gfx1250) — matrix operand order repaired and checked by reading, not compiled, not run on hardware
//
#include <hip/hip_runtime.h>
#include <stdint.h>

// Problem sizes (match reference)
constexpr int Bn  = 512;   // batch
constexpr int Tn  = 256;   // timesteps
constexpr int Hn  = 256;   // hidden
constexpr int Pn  = 14;    // predict dim
constexpr int G4H = 1024;  // 4*H

typedef __attribute__((ext_vector_type(16))) __bf16        v16bf;
typedef __attribute__((ext_vector_type(8)))  float         v8f;
typedef __attribute__((ext_vector_type(4)))  unsigned int  u32x4;

union Op16 { u32x4 u[2]; v16bf v; };  // two 16B chunks -> one 32B WMMA operand

__device__ __forceinline__ float sigf(float x)      { return 1.0f / (1.0f + __expf(-x)); }
__device__ __forceinline__ float tanh_fast(float x) { return 2.0f * sigf(2.0f * x) - 1.0f; }

// ---------------------------------------------------------------------------
// Prep: convert weights to bf16, packed so WMMA B-operand loads are
// lane-contiguous.  Wpk[col][k] == W^T[k][col] (W given row-major [4H,K]).
// Wpk1 concatenates K: cols 0..255 = w_ih1, 256..511 = w_hh1 (A = [h1_t | h2]).
// ---------------------------------------------------------------------------
__global__ __launch_bounds__(256) void pack_weights_kernel(
    const float* __restrict__ w_hh0, const float* __restrict__ w_ih1,
    const float* __restrict__ w_hh1, __bf16* __restrict__ Wpk0,
    __bf16* __restrict__ Wpk1) {
  int i = blockIdx.x * blockDim.x + threadIdx.x;
  if (i < G4H * Hn) Wpk0[i] = (__bf16)w_hh0[i];
  if (i < G4H * 2 * Hn) {
    int n = i >> 9, k = i & 511;
    float v = (k < Hn) ? w_ih1[n * Hn + k] : w_hh1[n * Hn + (k - Hn)];
    Wpk1[i] = (__bf16)v;
  }
}

// ---------------------------------------------------------------------------
// Fused LSTM layer: one workgroup = 16 batch rows, 8 waves.
// Wave w owns all 4 gates for hidden units [w*32, w*32+32): 8 f32 16x16 tiles.
// Per timestep: gates = Cinit + A[16 x K] * Wpk^T via v_wmma_f32_16x16x32_bf16,
// then pointwise LSTM update; h (bf16) goes back to LDS for the next step.
// LAYER==0: K=256 (A=h), pre computed inline from x (w_ih0 is [4H,1]).
// LAYER==1: K=512 (A=[h1_t | h2]), input GEMM fused into the recurrence.
// ---------------------------------------------------------------------------
template <int KDIM, int LAYER>
__global__ __launch_bounds__(256) void lstm_layer_kernel(
    const float*  __restrict__ x,          // [B,T]    (LAYER==0)
    const __bf16* __restrict__ h_in,       // [T,B,H]  (LAYER==1)
    __bf16*       __restrict__ h_out_seq,  // [T,B,H]  (LAYER==0)
    const float*  __restrict__ w_ih_vec,   // [4H]     (LAYER==0)
    const float*  __restrict__ b_ih,       // [4H]
    const float*  __restrict__ b_hh,       // [4H]
    const __bf16* __restrict__ Wpk,        // [4H][KDIM] packed bf16
    float*        __restrict__ h_final)    // [B,H]    (LAYER==1)
{
  __shared__ __align__(16) __bf16 hbuf[16][KDIM];
  __shared__ float xbuf[16];

  const int tid  = threadIdx.x;
  const int lane = tid & 31;
  const int hi   = lane >> 4;   // half-wave
  const int ln   = lane & 15;
  const int wave = tid >> 5;
  const int b0   = blockIdx.x * 16;
  const int ub   = wave * 32;   // hidden-unit base for this wave

  // h_{-1} = 0 (layer 1 only needs the recurrent half zeroed)
  for (int idx = tid; idx < 16 * KDIM; idx += 256) {
    int r = idx / KDIM, k = idx - r * KDIM;
    if (LAYER == 0 || k >= Hn) hbuf[r][k] = (__bf16)0.0f;
  }
  __syncthreads();

  // Per-lane constants for the 8 tiles this wave owns
  float bsum[4][2], wi0[4][2];
#pragma unroll
  for (int g = 0; g < 4; ++g)
#pragma unroll
    for (int s = 0; s < 2; ++s) {
      int col = g * Hn + ub + s * 16 + ln;
      bsum[g][s] = b_ih[col] + b_hh[col];
      wi0[g][s]  = (LAYER == 0) ? w_ih_vec[col] : 0.0f;
    }

  v8f cst[2];  // cell state, 16 rows x 32 units per wave
#pragma unroll
  for (int s = 0; s < 2; ++s)
#pragma unroll
    for (int r = 0; r < 8; ++r) cst[s][r] = 0.0f;

  for (int t = 0; t < Tn; ++t) {
    // ---- stage per-step inputs into LDS ----
    if (LAYER == 0) {
      if (tid < 16) xbuf[tid] = x[(size_t)(b0 + tid) * Tn + t];
    } else {
      const __bf16* src = h_in + ((size_t)t * Bn + b0) * Hn;
      int r = tid >> 4, seg = tid & 15;
      *(u32x4*)&hbuf[r][seg * 16]     = *(const u32x4*)(src + r * Hn + seg * 16);
      *(u32x4*)&hbuf[r][seg * 16 + 8] = *(const u32x4*)(src + r * Hn + seg * 16 + 8);
    }
    __syncthreads();  // staged inputs + previous step's h visible

    // ---- accumulator init: pre/bias in C/D layout (lane=N, vgpr r -> M) ----
    float xv[8];
    if (LAYER == 0) {
#pragma unroll
      for (int r = 0; r < 8; ++r) xv[r] = xbuf[hi * 8 + r];
    }
    v8f acc[4][2];
#pragma unroll
    for (int g = 0; g < 4; ++g)
#pragma unroll
      for (int s = 0; s < 2; ++s)
#pragma unroll
        for (int r = 0; r < 8; ++r)
          acc[g][s][r] = (LAYER == 0) ? fmaf(xv[r], wi0[g][s], bsum[g][s])
                                      : bsum[g][s];

    // ---- K loop: A from LDS (ISA A-interleave), B lane-contiguous from L2 ----
#pragma unroll 4
    for (int kc = 0; kc < KDIM; kc += 32) {
      Op16 A;
      A.u[0] = *(const u32x4*)&hbuf[ln][kc + hi * 8];
      A.u[1] = *(const u32x4*)&hbuf[ln][kc + 16 + hi * 8];
#pragma unroll
      for (int g = 0; g < 4; ++g)
#pragma unroll
        for (int s = 0; s < 2; ++s) {
          const __bf16* bp =
              Wpk + (size_t)(g * Hn + ub + s * 16 + ln) * KDIM + kc + hi * 8;
          Op16 Bv;
          Bv.u[0] = *(const u32x4*)bp;
          Bv.u[1] = *(const u32x4*)(bp + 16);
          acc[g][s] = __builtin_amdgcn_wmma_f32_16x16x32_bf16(
              false, A.v, false, Bv.v, (short)0, acc[g][s], false, false);
        }
    }
    __syncthreads();  // all reads of hbuf done before h overwrite

    // ---- pointwise LSTM update; write new h (bf16) back to LDS ----
#pragma unroll
    for (int s = 0; s < 2; ++s) {
      int col = ub + s * 16 + ln;
#pragma unroll
      for (int r = 0; r < 8; ++r) {
        float iv = sigf(acc[0][s][r]);
        float fv = sigf(acc[1][s][r]);
        float gv = tanh_fast(acc[2][s][r]);
        float ov = sigf(acc[3][s][r]);
        float c  = fv * cst[s][r] + iv * gv;
        cst[s][r] = c;
        float h  = ov * tanh_fast(c);
        int row  = hi * 8 + r;
        __bf16 hb = (__bf16)h;
        hbuf[row][(LAYER ? Hn : 0) + col] = hb;
        if (LAYER == 0)
          h_out_seq[((size_t)t * Bn + b0 + row) * Hn + col] = hb;
      }
    }
    // next iteration's top barrier publishes these writes
  }

  if (LAYER == 1) {
    __syncthreads();
    for (int idx = tid; idx < 16 * Hn; idx += 256) {
      int r = idx / Hn, k = idx & (Hn - 1);
      h_final[(size_t)(b0 + r) * Hn + k] = (float)hbuf[r][Hn + k];
    }
  }
}

// ---------------------------------------------------------------------------
// Tiny final projection: out[b,p] = h2[b,:] . w_lin[p,:] + b_lin[p]
// ---------------------------------------------------------------------------
__global__ __launch_bounds__(256) void final_linear_kernel(
    const float* __restrict__ h2, const float* __restrict__ w_lin,
    const float* __restrict__ b_lin, float* __restrict__ out) {
  int i = blockIdx.x * blockDim.x + threadIdx.x;
  if (i >= Bn * Pn) return;
  int b = i / Pn, p = i - b * Pn;
  const float4* hv = (const float4*)(h2 + (size_t)b * Hn);
  const float4* wv = (const float4*)(w_lin + (size_t)p * Hn);
  float acc = b_lin[p];
#pragma unroll 8
  for (int k = 0; k < Hn / 4; ++k) {
    float4 a = hv[k], w = wv[k];
    acc += a.x * w.x + a.y * w.y + a.z * w.z + a.w * w.w;
  }
  out[i] = acc;
}

// ---------------------------------------------------------------------------
extern "C" void kernel_launch(void* const* d_in, const int* in_sizes, int n_in,
                              void* d_out, int out_size, void* d_ws,
                              size_t ws_size, hipStream_t stream) {
  const float* x     = (const float*)d_in[0];
  const float* w_ih0 = (const float*)d_in[1];
  const float* w_hh0 = (const float*)d_in[2];
  const float* b_ih0 = (const float*)d_in[3];
  const float* b_hh0 = (const float*)d_in[4];
  const float* w_ih1 = (const float*)d_in[5];
  const float* w_hh1 = (const float*)d_in[6];
  const float* b_ih1 = (const float*)d_in[7];
  const float* b_hh1 = (const float*)d_in[8];
  const float* w_lin = (const float*)d_in[9];
  const float* b_lin = (const float*)d_in[10];
  float* out = (float*)d_out;

  // Workspace carve (~66 MB): packed weights + h1 sequence + final h2
  char* ws = (char*)d_ws;
  size_t off = 0;
  __bf16* Wpk0 = (__bf16*)(ws + off); off += (size_t)G4H * Hn * 2;      // 512 KB
  __bf16* Wpk1 = (__bf16*)(ws + off); off += (size_t)G4H * 2 * Hn * 2;  // 1 MB
  __bf16* h1   = (__bf16*)(ws + off); off += (size_t)Tn * Bn * Hn * 2;  // 64 MB
  float*  h2f  = (float*)(ws + off);  off += (size_t)Bn * Hn * 4;       // 512 KB
  (void)ws_size; (void)in_sizes; (void)n_in; (void)out_size;

  pack_weights_kernel<<<(G4H * 2 * Hn + 255) / 256, 256, 0, stream>>>(
      w_hh0, w_ih1, w_hh1, Wpk0, Wpk1);

  lstm_layer_kernel<Hn, 0><<<Bn / 16, 256, 0, stream>>>(
      x, nullptr, h1, w_ih0, b_ih0, b_hh0, Wpk0, nullptr);

  lstm_layer_kernel<2 * Hn, 1><<<Bn / 16, 256, 0, stream>>>(
      nullptr, h1, nullptr, nullptr, b_ih1, b_hh1, Wpk1, h2f);

  final_linear_kernel<<<(Bn * Pn + 255) / 256, 256, 0, stream>>>(
      h2f, w_lin, b_lin, out);
}
